// Encoder_34479997453173
// MI455X (gfx1250) — matrix operand order repaired and checked by reading, not compiled, not run on hardware
//
#include <hip/hip_runtime.h>
#include <stdint.h>

// ---------------- problem constants ----------------
#define B_   128
#define T_   4096
#define F_   64
#define H_   128
#define BT   16        // batch tile per workgroup (WMMA M)
#define NTH  256       // 8 waves of 32; wave w owns hidden columns [16w,16w+16)

// LDS strides (elements), padded so the 16 rows spread across banks
#define H_STRIDE   136     // bf16 row of h tile (128 + 8 pad)
#define X_STRIDE   72      // bf16 row of x tile (64 + 8 pad)

typedef __attribute__((ext_vector_type(16))) __bf16 v16bf;
typedef __attribute__((ext_vector_type(8)))  float  v8f;

// ---------------- helpers ----------------
__device__ __forceinline__ unsigned short bf16_bits(float x) {
    unsigned u = __builtin_bit_cast(unsigned, x);
    return (unsigned short)((u + 0x7FFFu + ((u >> 16) & 1u)) >> 16);   // RNE
}
__device__ __forceinline__ __bf16 f2bf(float x) {
    return __builtin_bit_cast(__bf16, bf16_bits(x));
}
__device__ __forceinline__ unsigned pack2bf(float a, float b) {
    return (unsigned)bf16_bits(a) | ((unsigned)bf16_bits(b) << 16);
}

// CDNA5 has V_TANH_F32: 1 TRANS op for tanh, and sigmoid via
// sig(x) = 0.5*tanh(x/2) + 0.5  (1 TRANS + 1 FMA). Fall back to exp form.
#if __has_builtin(__builtin_amdgcn_tanhf)
__device__ __forceinline__ float ftanh(float x) { return __builtin_amdgcn_tanhf(x); }
#elif __has_builtin(__builtin_amdgcn_tanh_f32)
__device__ __forceinline__ float ftanh(float x) { return __builtin_amdgcn_tanh_f32(x); }
#else
__device__ __forceinline__ float ftanh(float x) { return 2.0f / (1.0f + __expf(-2.0f * x)) - 1.0f; }
#endif
__device__ __forceinline__ float fsig(float x)  { return __builtin_fmaf(0.5f, ftanh(0.5f * x), 0.5f); }

union FragU { uint4 q[2]; unsigned u[8]; v16bf v; };
union AccU  { v8f v; float f[8]; };

// Per-lane B fragment (32x16 bf16): 16 contiguous K of W row n, converted from f32.
__device__ __forceinline__ v16bf load_wfrag(const float* p) {
    const float4* q = (const float4*)p;
    float4 a = q[0], b = q[1], c = q[2], d = q[3];
    FragU r;
    r.u[0] = pack2bf(a.x, a.y); r.u[1] = pack2bf(a.z, a.w);
    r.u[2] = pack2bf(b.x, b.y); r.u[3] = pack2bf(b.z, b.w);
    r.u[4] = pack2bf(c.x, c.y); r.u[5] = pack2bf(c.z, c.w);
    r.u[6] = pack2bf(d.x, d.y); r.u[7] = pack2bf(d.z, d.w);
    return r.v;
}
// Per-lane A fragment (16x32 bf16) from row-major LDS:
// lanes 0-15: K {kb..kb+15}; lanes 16-31: K {kb+16..kb+31}.
__device__ __forceinline__ v16bf load_afrag(const __bf16* p) {
    FragU u;
    u.q[0] = *(const uint4*)(p);
    u.q[1] = *(const uint4*)(p + 8);
    return u.v;
}

// ---------------- kernel ----------------
__global__ __launch_bounds__(NTH)
void lstm_persistent_kernel(const float* __restrict__ X,
                            const float* __restrict__ W_ih,
                            const float* __restrict__ W_hh,
                            const float* __restrict__ b_ih,
                            const float* __restrict__ b_hh,
                            float* __restrict__ out)
{
    __shared__ __bf16 hbuf[2][BT * H_STRIDE];   // parity t&1
    __shared__ __bf16 xbuf[2][BT * X_STRIDE];   // parity t&1

    const int tid  = threadIdx.x;
    const int lane = tid & 31;
    const int wave = tid >> 5;          // 0..7
    const int row  = lane & 15;         // N within tile / A row group
    const int half = lane >> 4;         // K-half (A/B) == M-half (C/D)
    const int b0   = blockIdx.x * BT;
    const int jcol = wave * 16 + row;   // this lane's hidden column j (0..127)

    // ---- one-time: weights f32 -> bf16 register fragments (4096x reuse) ----
    // Wave w computes gate tiles {i,f,g,o} x columns [16w,16w+16):
    // B row n = G*128 + jcol;  K block = kf*32 + half*16 (16 contiguous floats).
    v16bf whhB[4][4];   // 128 VGPRs
    v16bf wihB[4][2];   //  64 VGPRs
    v8f   biasv[4];     // bias[G*128 + jcol] broadcast across the 8 acc slots:
                        // used as initial C of each WMMA chain (free bias add)
    #pragma unroll
    for (int G = 0; G < 4; ++G) {
        const int n = G * 128 + jcol;
        #pragma unroll
        for (int kf = 0; kf < 4; ++kf)
            whhB[G][kf] = load_wfrag(W_hh + (size_t)n * H_ + kf * 32 + half * 16);
        #pragma unroll
        for (int kf = 0; kf < 2; ++kf)
            wihB[G][kf] = load_wfrag(W_ih + (size_t)n * F_ + kf * 32 + half * 16);
        const float bsum = b_ih[n] + b_hh[n];
        AccU bu;
        #pragma unroll
        for (int r = 0; r < 8; ++r) bu.f[r] = bsum;
        biasv[G] = bu.v;
    }

    // zero initial hidden state buffers
    for (int i = tid; i < 2 * BT * H_STRIDE; i += NTH) hbuf[0][i] = f2bf(0.0f);

    // cell state: lane holds c[m][jcol] for m = 8*half + r, r=0..7 (registers forever)
    float cst[8];
    #pragma unroll
    for (int r = 0; r < 8; ++r) cst[r] = 0.0f;

    // register double-buffer for the X tile: thread -> (m = tid/16, f4 = 4*(tid%16))
    const int xm  = tid >> 4;
    const int xf4 = (tid & 15) << 2;
    const float* xbase = X + ((size_t)(b0 + xm) * T_) * F_ + xf4;
    float4 xv = *(const float4*)(xbase);          // t = 0
    __syncthreads();

    // ---- recurrence over T: ONE barrier per step ----
    for (int t = 0; t < T_; ++t) {
        // stage x(t) into LDS parity buffer; start load of x(t+1); prefetch x(t+2)
        {
            uint2 p; p.x = pack2bf(xv.x, xv.y); p.y = pack2bf(xv.z, xv.w);
            *(uint2*)(&xbuf[t & 1][xm * X_STRIDE + xf4]) = p;
        }
        float4 xnext = xv;
        if (t + 1 < T_) {
            const float* xp = xbase + (size_t)(t + 1) * F_;
            xnext = *(const float4*)xp;
            __builtin_prefetch(xp + F_, 0, 3);
        }
        __syncthreads();   // x(t) staged; h(t) from previous step visible

        // A fragments (shared by all 4 gate chains of this wave)
        const __bf16* hrow = &hbuf[t & 1][row * H_STRIDE] + half * 16;
        const __bf16* xrow = &xbuf[t & 1][row * X_STRIDE] + half * 16;
        v16bf ha[4], xa[2];
        #pragma unroll
        for (int kf = 0; kf < 4; ++kf) ha[kf] = load_afrag(hrow + kf * 32);
        #pragma unroll
        for (int kf = 0; kf < 2; ++kf) xa[kf] = load_afrag(xrow + kf * 32);

        // 4 independent 6-deep WMMA chains (i,f,g,o) -> hazard-free interleave.
        // Chain seeded with broadcast bias as C: bias add costs zero VALU.
        v8f acc[4];
        #pragma unroll
        for (int G = 0; G < 4; ++G)
            acc[G] = __builtin_amdgcn_wmma_f32_16x16x32_bf16(
                        false, ha[0], false, whhB[G][0], (short)0, biasv[G], false, false);
        #pragma unroll
        for (int kf = 1; kf < 4; ++kf)
            #pragma unroll
            for (int G = 0; G < 4; ++G)
                acc[G] = __builtin_amdgcn_wmma_f32_16x16x32_bf16(
                            false, ha[kf], false, whhB[G][kf], (short)0, acc[G], false, false);
        #pragma unroll
        for (int kf = 0; kf < 2; ++kf)
            #pragma unroll
            for (int G = 0; G < 4; ++G)
                acc[G] = __builtin_amdgcn_wmma_f32_16x16x32_bf16(
                            false, xa[kf], false, wihB[G][kf], (short)0, acc[G], false, false);

        // fused LSTM cell update, fully in-register:
        // lane owns (m = 8*half + r, j = jcol) for r=0..7 in ALL four accumulators
        AccU ai, af, ag, ao;
        ai.v = acc[0]; af.v = acc[1]; ag.v = acc[2]; ao.v = acc[3];
        __bf16* hw = &hbuf[(t + 1) & 1][0];
        #pragma unroll
        for (int r = 0; r < 8; ++r) {
            const int m = 8 * half + r;
            float iv = fsig (ai.f[r]);
            float fv = fsig (af.f[r]);
            float gv = ftanh(ag.f[r]);
            float ov = fsig (ao.f[r]);
            float c  = fv * cst[r] + iv * gv;
            cst[r] = c;
            float h = ov * ftanh(c);
            hw[m * H_STRIDE + jcol] = f2bf(h);
            if (t == T_ - 1) out[(size_t)(b0 + m) * H_ + jcol] = h;
        }
        xv = xnext;
        // next iteration's barrier orders these h stores before their readers
    }
}

// ---------------- launcher ----------------
extern "C" void kernel_launch(void* const* d_in, const int* in_sizes, int n_in,
                              void* d_out, int out_size, void* d_ws, size_t ws_size,
                              hipStream_t stream) {
    (void)in_sizes; (void)n_in; (void)out_size; (void)d_ws; (void)ws_size;
    const float* X    = (const float*)d_in[0];
    const float* W_ih = (const float*)d_in[1];
    const float* W_hh = (const float*)d_in[2];
    const float* b_ih = (const float*)d_in[3];
    const float* b_hh = (const float*)d_in[4];
    float* out = (float*)d_out;

    dim3 grid(B_ / BT);   // 8 independent workgroups, one batch tile each
    dim3 block(NTH);      // 8 wave32 waves; wave w owns hidden cols [16w,16w+16)
    lstm_persistent_kernel<<<grid, block, 0, stream>>>(X, W_ih, W_hh, b_ih, b_hh, out);
}
